// MGF_74741020885552
// MI455X (gfx1250) — matrix operand order repaired and checked by reading, not compiled, not run on hardware
//
#include <hip/hip_runtime.h>
#include <math.h>

typedef __bf16 bf16;
typedef __attribute__((ext_vector_type(16))) __bf16 v16bf;
typedef __attribute__((ext_vector_type(8)))  float   v8f;

constexpr int HH  = 256;
constexpr int WWD = 256;
constexpr int CC  = 64;
constexpr int BB  = 2;
constexpr int HWP = HH * WWD;          // 65536 pixels per image
constexpr int NPIX = BB * HWP;         // 131072 total pixels

union Frag {
  v16bf v;
  unsigned int u[8];
};

// ---------------------------------------------------------------------------
// Pack f32 OIHW conv weights into WMMA A-fragment order (bf16).
// dst layout: [tap][ks(0..1)][mf][lane(0..31)][elem(0..15)]
// A 16x32 bf16 fragment in this code's contiguous k order (B loads match):
//   lane<16 : M=lane,    elems 0..15 -> K 0..15
//   lane>=16: M=lane-16, elems 0..15 -> K 16..31
// ---------------------------------------------------------------------------
__global__ void pack_w(const float* __restrict__ w, bf16* __restrict__ dst,
                       int ntaps, int mfrags, int cout_valid, int total) {
  int idx = blockIdx.x * 256 + threadIdx.x;
  if (idx >= total) return;
  int e    = idx & 15;
  int lane = (idx >> 4) & 31;
  int rest = idx >> 9;
  int mf = rest % mfrags; rest /= mfrags;
  int ks = rest & 1;
  int tap = rest >> 1;
  int cout = mf * 16 + (lane & 15);
  int klocal = e + ((lane >> 4) << 4);
  int cin = ks * 32 + klocal;
  float val = 0.0f;
  if (cout < cout_valid) {
    if (ntaps == 9) val = w[(cout * CC + cin) * 9 + tap];
    else            val = w[cout * CC + cin];
  }
  dst[idx] = (bf16)val;
}

// NCHW f32 -> NHWC bf16
__global__ void to_nhwc(const float* __restrict__ src, bf16* __restrict__ dst) {
  size_t i = (size_t)blockIdx.x * 256 + threadIdx.x;
  int c = (int)(i & 63);
  size_t p = (i >> 6) & (size_t)(HWP - 1);
  size_t b = i >> 22;
  dst[i] = (bf16)src[(b * CC + c) * HWP + p];
}

// ---------------------------------------------------------------------------
// WMMA implicit-GEMM conv (3x3 via 9 shifted taps, or 1x1).
// in/out NHWC bf16.  Block = 8 waves x 32 pixels = one full W row (256 px).
// Each wave: M = 16*MFRAGS couts, N = 32 pixels (2 N-tiles sharing A frags).
// Weights staged to LDS with GLOBAL_LOAD_ASYNC_TO_LDS (ASYNCcnt path).
// act: 0=none 1=LeakyReLU(0.2) 2=ReLU.  resid: optional NHWC bf16 add.
// out_mode: 0=bf16 NHWC, 1=f32 NCHW, 2=f32 [pix][9]
// ---------------------------------------------------------------------------
template <int NTAPS, int MFRAGS>
__global__ __launch_bounds__(256) void conv_wmma(
    const bf16* __restrict__ in, const bf16* __restrict__ wpack,
    const float* __restrict__ bias, const bf16* __restrict__ resid,
    bf16* __restrict__ out_bf, float* __restrict__ out_f32,
    int act, int out_mode, int cout_limit) {
  constexpr int TC = (NTAPS > 5) ? 5 : NTAPS;      // LDS stage: <= 40KB
  __shared__ __align__(16) bf16 sw[TC * 2 * MFRAGS * 512];

  const int tid   = threadIdx.x;
  const int wv    = tid >> 5;        // wave id (wave32)
  const int lane  = tid & 31;
  const int colN  = lane & 15;       // N column within fragment
  const int khalf = lane >> 4;       // which 16-channel half this lane loads

  const int bid = blockIdx.x;        // 512 blocks: (b, h)
  const int b   = bid >> 8;
  const int h   = bid & 255;
  const int wx  = wv * 32 + colN;    // first N-tile pixel x (second: +16)

  v8f acc[MFRAGS][2];
  #pragma unroll
  for (int m = 0; m < MFRAGS; ++m)
    #pragma unroll
    for (int nt = 0; nt < 2; ++nt)
      #pragma unroll
      for (int j = 0; j < 8; ++j) acc[m][nt][j] = 0.0f;

  for (int tap0 = 0; tap0 < NTAPS; tap0 += TC) {
    const int tcnt = (NTAPS - tap0 < TC) ? (NTAPS - tap0) : TC;
    __syncthreads();   // previous stage fully consumed before overwrite
    {
      // Async DMA: global weights -> LDS, 16B per lane per sweep.
      const int bytes = tcnt * 2 * MFRAGS * 512 * (int)sizeof(bf16);
      const char* gsrc = (const char*)wpack + (size_t)tap0 * 2 * MFRAGS * 512 * sizeof(bf16);
      const unsigned lbase = (unsigned)(size_t)(&sw[0]);
      for (int ofs = tid * 16; ofs < bytes; ofs += 4096) {
        unsigned laddr = lbase + (unsigned)ofs;
        const char* ga = gsrc + ofs;
        asm volatile("global_load_async_to_lds_b128 %0, %1, off"
                     :: "v"(laddr), "v"(ga) : "memory");
      }
      asm volatile("s_wait_asynccnt 0x0" ::: "memory");
    }
    __syncthreads();

    for (int t = 0; t < tcnt; ++t) {
      const int tap = tap0 + t;
      int di = 0, dj = 0;
      if (NTAPS == 9) { di = tap / 3 - 1; dj = tap % 3 - 1; }
      const int py = h + di;
      Frag bt[2][2];   // [ntile][kstep]
      #pragma unroll
      for (int nt = 0; nt < 2; ++nt) {
        const int px = wx + nt * 16 + dj;
        #pragma unroll
        for (int j = 0; j < 8; ++j) { bt[nt][0].u[j] = 0u; bt[nt][1].u[j] = 0u; }
        if (py >= 0 && py < HH && px >= 0 && px < WWD) {
          const bf16* bp = in + ((size_t)b * HWP + (size_t)py * WWD + px) * CC + khalf * 16;
          bt[nt][0].v = *(const v16bf*)bp;          // channels 0..31 half
          bt[nt][1].v = *(const v16bf*)(bp + 32);   // channels 32..63 half
        }
      }
      #pragma unroll
      for (int mf = 0; mf < MFRAGS; ++mf) {
        const v16bf a0 = *(const v16bf*)&sw[(size_t)(((t * 2 + 0) * MFRAGS + mf) * 32 + lane) * 16];
        const v16bf a1 = *(const v16bf*)&sw[(size_t)(((t * 2 + 1) * MFRAGS + mf) * 32 + lane) * 16];
        #pragma unroll
        for (int nt = 0; nt < 2; ++nt) {
          acc[mf][nt] = __builtin_amdgcn_wmma_f32_16x16x32_bf16(false, a0, false, bt[nt][0].v,
                                                                (short)0, acc[mf][nt], false, false);
          acc[mf][nt] = __builtin_amdgcn_wmma_f32_16x16x32_bf16(false, a1, false, bt[nt][1].v,
                                                                (short)0, acc[mf][nt], false, false);
        }
      }
    }
  }

  // Epilogue.  C layout: lanes 0-15: N=lane, M=vgpr; lanes 16-31: M=vgpr+8.
  #pragma unroll
  for (int nt = 0; nt < 2; ++nt) {
    const int wxn = wx + nt * 16;
    const size_t pidx = (size_t)b * HWP + (size_t)h * WWD + wxn;
    #pragma unroll
    for (int mf = 0; mf < MFRAGS; ++mf) {
      #pragma unroll
      for (int v = 0; v < 8; ++v) {
        const int cout = mf * 16 + v + khalf * 8;
        float val = acc[mf][nt][v] + ((cout < cout_limit) ? bias[cout] : 0.0f);
        if (act == 1)      val = (val > 0.0f) ? val : 0.2f * val;
        else if (act == 2) val = (val > 0.0f) ? val : 0.0f;
        if (resid) val += (float)resid[pidx * CC + cout];
        if (out_mode == 0) {
          out_bf[pidx * CC + cout] = (bf16)val;
        } else if (out_mode == 1) {
          out_f32[((size_t)(b * CC + cout)) * HWP + (size_t)h * WWD + wxn] = val;
        } else {
          if (cout < cout_limit) out_f32[pidx * 9 + cout] = val;
        }
      }
    }
  }
}

// Channel attention: mean over HxW per (b,c)
__global__ __launch_bounds__(256) void ca_reduce(const bf16* __restrict__ x,
                                                 float* __restrict__ mean) {
  const int bc = blockIdx.x;
  const int b = bc >> 6, c = bc & 63;
  const bf16* p = x + (size_t)b * HWP * CC + c;
  float s = 0.0f;
  for (int i = threadIdx.x; i < HWP; i += 256) s += (float)p[(size_t)i * CC];
  __shared__ float red[256];
  red[threadIdx.x] = s;
  __syncthreads();
  for (int st = 128; st > 0; st >>= 1) {
    if (threadIdx.x < st) red[threadIdx.x] += red[threadIdx.x + st];
    __syncthreads();
  }
  if (threadIdx.x == 0) mean[bc] = red[0] * (1.0f / 65536.0f);
}

// Channel attention MLP: scale = sigmoid(W2 * relu(W1 * mean + b1) + b2)
__global__ void ca_mlp(const float* __restrict__ mean,
                       const float* __restrict__ w1, const float* __restrict__ b1,
                       const float* __restrict__ w2, const float* __restrict__ b2,
                       float* __restrict__ scale) {
  const int b = blockIdx.x, t = threadIdx.x;   // 64 threads
  __shared__ float m[64];
  __shared__ float h1[4];
  m[t] = mean[b * 64 + t];
  __syncthreads();
  if (t < 4) {
    float s = b1[t];
    for (int i = 0; i < 64; ++i) s += w1[t * 64 + i] * m[i];
    h1[t] = (s > 0.0f) ? s : 0.0f;
  }
  __syncthreads();
  float s = b2[t];
  for (int i = 0; i < 4; ++i) s += w2[t * 4 + i] * h1[i];
  scale[b * 64 + t] = 1.0f / (1.0f + __expf(-s));
}

// r4 = r2 * scale[b,c] + d   (NHWC bf16)
__global__ void scale_add(const bf16* __restrict__ r2, const float* __restrict__ scale,
                          const bf16* __restrict__ d, bf16* __restrict__ out) {
  size_t i = (size_t)blockIdx.x * 256 + threadIdx.x;
  int c = (int)(i & 63);
  int b = (int)(i >> 22);
  out[i] = (bf16)((float)r2[i] * scale[b * CC + c] + (float)d[i]);
}

// Fused: t = raw*(1+S); o = W9x9*t + b; f = tanh(o)/(aff+1e-8); kern = f/max(sum|f|+1e-4,1)
__global__ void make_kernel9(const float* __restrict__ raw, const float* __restrict__ S,
                             const float* __restrict__ w, const float* __restrict__ bias,
                             const float* __restrict__ aff, float* __restrict__ kern) {
  int p = blockIdx.x * 256 + threadIdx.x;
  if (p >= NPIX) return;
  const float sv = 1.0f + S[p];
  float t[9], o[9];
  #pragma unroll
  for (int k = 0; k < 9; ++k) t[k] = raw[(size_t)p * 9 + k] * sv;
  const float afv = aff[0] + 1e-8f;
  float ssum = 0.0f;
  #pragma unroll
  for (int ko = 0; ko < 9; ++ko) {
    float a = bias[ko];
    #pragma unroll
    for (int ki = 0; ki < 9; ++ki) a += w[ko * 9 + ki] * t[ki];
    float f = tanhf(a) / afv;
    o[ko] = f;
    ssum += fabsf(f);
  }
  ssum += 1e-4f;
  if (ssum < 1.0f) ssum = 1.0f;
  const float inv = 1.0f / ssum;
  #pragma unroll
  for (int k = 0; k < 9; ++k) kern[(size_t)p * 9 + k] = o[k] * inv;
}

// out[b,:,h,w] = sum_k kern[b,k,h,w] * inp[b,:,h+di,w+dj] + inp[b,:,h,w]
__global__ __launch_bounds__(256) void einsum9(const float* __restrict__ kern,
                                               const bf16* __restrict__ inp,
                                               bf16* __restrict__ out) {
  int idx = blockIdx.x * 256 + threadIdx.x;
  if (idx >= NPIX * 4) return;
  const int cg = idx & 3;          // 16-channel group
  const int p  = idx >> 2;
  const int b  = p >> 16;
  const int hw = p & (HWP - 1);
  const int h  = hw >> 8;
  const int w  = hw & 255;
  float kv[9];
  #pragma unroll
  for (int k = 0; k < 9; ++k) kv[k] = kern[(size_t)p * 9 + k];
  float accf[16];
  {
    Frag c0; c0.v = *(const v16bf*)(inp + (size_t)p * CC + cg * 16);
    #pragma unroll
    for (int j = 0; j < 16; ++j) accf[j] = (float)c0.v[j];
  }
  #pragma unroll
  for (int k = 0; k < 9; ++k) {
    const int py = h + k / 3 - 1;
    const int px = w + k % 3 - 1;
    if (py < 0 || py >= HH || px < 0 || px >= WWD) continue;
    Frag n;
    n.v = *(const v16bf*)(inp + ((size_t)b * HWP + (size_t)py * WWD + px) * CC + cg * 16);
    #pragma unroll
    for (int j = 0; j < 16; ++j) accf[j] += kv[k] * (float)n.v[j];
  }
  bf16* op = out + (size_t)p * CC + cg * 16;
  #pragma unroll
  for (int j = 0; j < 16; ++j) op[j] = (bf16)accf[j];
}

// ---------------------------------------------------------------------------
extern "C" void kernel_launch(void* const* d_in, const int* in_sizes, int n_in,
                              void* d_out, int out_size, void* d_ws, size_t ws_size,
                              hipStream_t stream) {
  (void)in_sizes; (void)n_in; (void)out_size; (void)ws_size;

  #define PF(i) ((const float*)d_in[(i)])
  const float* depth = PF(0);
  const float* guide = PF(1);
  const float* S     = PF(2);
  const float* aff   = PF(3);
  const float *d1w = PF(4),  *d1b = PF(5),  *g1w = PF(6),  *g1b = PF(7);
  const float *dk1w = PF(8), *dk1b = PF(9), *dk2w = PF(10), *dk2b = PF(11);
  const float *gk1w = PF(12),*gk1b = PF(13),*gk2w = PF(14), *gk2b = PF(15);
  const float *d2w = PF(16), *d2b = PF(17), *g2w = PF(18),  *g2b = PF(19);
  const float *d3w = PF(20), *d3b = PF(21), *g3w = PF(22),  *g3b = PF(23);
  const float *ic_c1w = PF(24), *ic_c1b = PF(25), *ic_c2w = PF(26), *ic_c2b = PF(27);
  const float *ic_ca1w = PF(28), *ic_ca1b = PF(29), *ic_ca2w = PF(30), *ic_ca2b = PF(31);
  const float *ic_tw = PF(32), *ic_tb = PF(33);
  const float *gc_c1w = PF(34), *gc_c1b = PF(35), *gc_c2w = PF(36), *gc_c2b = PF(37);
  const float *gc_ca1w = PF(38), *gc_ca1b = PF(39), *gc_ca2w = PF(40), *gc_ca2b = PF(41);
  const float *gc_tw = PF(42), *gc_tb = PF(43);
  #undef PF

  // ---- workspace carving ----
  char* base = (char*)d_ws;
  size_t off = 0;
  auto carve = [&](size_t bytes) -> char* {
    char* p = base + off;
    off += (bytes + 255) & ~(size_t)255;
    return p;
  };
  const size_t NBF = (size_t)NPIX * CC * sizeof(bf16);   // 16 MB per activation
  bf16* S0 = (bf16*)carve(NBF);   // depth(bf16) -> later w_depth
  bf16* S1 = (bf16*)carve(NBF);   // guide(bf16) -> later w_guide
  bf16* S2 = (bf16*)carve(NBF);   // d
  bf16* S3 = (bf16*)carve(NBF);   // g
  bf16* S4 = (bf16*)carve(NBF);   // r1 / gk1-out / dk1-out (scratch)
  bf16* S5 = (bf16*)carve(NBF);   // r2
  bf16* S6 = (bf16*)carve(NBF);   // r4
  bf16* S7 = (bf16*)carve(NBF);   // inp_d -> later inp_g
  float* Fraw  = (float*)carve((size_t)NPIX * 9 * 4);
  float* Fkern = (float*)carve((size_t)NPIX * 9 * 4);
  float* camean  = (float*)carve(BB * CC * 4);
  float* cascale = (float*)carve(BB * CC * 4);
  const size_t PK3 = 9 * 2 * 4 * 512 * sizeof(bf16);   // 72 KB
  bf16* pk_d1   = (bf16*)carve(PK3);
  bf16* pk_g1   = (bf16*)carve(PK3);
  bf16* pk_ic1  = (bf16*)carve(PK3);
  bf16* pk_ic2  = (bf16*)carve(PK3);
  bf16* pk_ict  = (bf16*)carve(PK3);
  bf16* pk_gc1  = (bf16*)carve(PK3);
  bf16* pk_gc2  = (bf16*)carve(PK3);
  bf16* pk_gct  = (bf16*)carve(PK3);
  bf16* pk_d3   = (bf16*)carve(PK3);
  bf16* pk_g3   = (bf16*)carve(PK3);
  bf16* pk_gk1  = (bf16*)carve(2 * 4 * 512 * sizeof(bf16));
  bf16* pk_dk1  = (bf16*)carve(2 * 4 * 512 * sizeof(bf16));
  bf16* pk_gk2  = (bf16*)carve(2 * 1 * 512 * sizeof(bf16));
  bf16* pk_dk2  = (bf16*)carve(2 * 1 * 512 * sizeof(bf16));

  // ---- launch helpers ----
  auto pack3 = [&](const float* w, bf16* dst) {
    const int total = 9 * 2 * 4 * 512;
    pack_w<<<(total + 255) / 256, 256, 0, stream>>>(w, dst, 9, 4, 64, total);
  };
  auto pack1 = [&](const float* w, bf16* dst, int mfrags, int cvalid) {
    const int total = 2 * mfrags * 512;
    pack_w<<<(total + 255) / 256, 256, 0, stream>>>(w, dst, 1, mfrags, cvalid, total);
  };
  auto conv9 = [&](const bf16* in, const bf16* wpk, const float* bias, const bf16* resid,
                   bf16* ob, float* of, int act, int omode) {
    conv_wmma<9, 4><<<512, 256, 0, stream>>>(in, wpk, bias, resid, ob, of, act, omode, 64);
  };
  auto conv1_64 = [&](const bf16* in, const bf16* wpk, const float* bias, bf16* ob, int act) {
    conv_wmma<1, 4><<<512, 256, 0, stream>>>(in, wpk, bias, nullptr, ob, nullptr, act, 0, 64);
  };
  auto conv1_9 = [&](const bf16* in, const bf16* wpk, const float* bias, float* of) {
    conv_wmma<1, 1><<<512, 256, 0, stream>>>(in, wpk, bias, nullptr, nullptr, of, 0, 2, 9);
  };
  auto res_group = [&](const bf16* x, bf16* outbuf,
                       const bf16* pkc1, const float* c1b, const bf16* pkc2, const float* c2b,
                       const float* ca1w, const float* ca1b, const float* ca2w, const float* ca2b,
                       const bf16* pkt, const float* tb) {
    conv9(x, pkc1, c1b, nullptr, S4, nullptr, 1, 0);             // r1 = leaky(conv)
    conv9(S4, pkc2, c2b, nullptr, S5, nullptr, 0, 0);            // r2
    ca_reduce<<<BB * CC, 256, 0, stream>>>(S5, camean);
    ca_mlp<<<BB, 64, 0, stream>>>(camean, ca1w, ca1b, ca2w, ca2b, cascale);
    scale_add<<<(int)((size_t)NPIX * CC / 256), 256, 0, stream>>>(S5, cascale, x, S6); // r4
    conv9(S6, pkt, tb, x, outbuf, nullptr, 0, 0);                // tail + residual
  };

  // ---- 1. weight packing ----
  pack3(d1w, pk_d1);   pack3(g1w, pk_g1);
  pack3(ic_c1w, pk_ic1); pack3(ic_c2w, pk_ic2); pack3(ic_tw, pk_ict);
  pack3(gc_c1w, pk_gc1); pack3(gc_c2w, pk_gc2); pack3(gc_tw, pk_gct);
  pack3(d3w, pk_d3);   pack3(g3w, pk_g3);
  pack1(gk1w, pk_gk1, 4, 64); pack1(dk1w, pk_dk1, 4, 64);
  pack1(gk2w, pk_gk2, 1, 9);  pack1(dk2w, pk_dk2, 1, 9);

  // ---- 2. input layout conversion ----
  to_nhwc<<<(int)((size_t)NPIX * CC / 256), 256, 0, stream>>>(depth, S0);
  to_nhwc<<<(int)((size_t)NPIX * CC / 256), 256, 0, stream>>>(guide, S1);

  // ---- 3. head convs ----
  conv9(S0, pk_d1, d1b, nullptr, S2, nullptr, 0, 0);   // d
  conv9(S1, pk_g1, g1b, nullptr, S3, nullptr, 0, 0);   // g

  // ---- 4. inp_d = res_group(d, ic) ----
  res_group(S2, S7, pk_ic1, ic_c1b, pk_ic2, ic_c2b,
            ic_ca1w, ic_ca1b, ic_ca2w, ic_ca2b, pk_ict, ic_tb);

  // ---- 5. guide-kernel branch -> kern ----
  conv1_64(S3, pk_gk1, gk1b, S4, 2);                   // relu(1x1)
  conv1_9(S4, pk_gk2, gk2b, Fraw);                     // 9-ch raw
  make_kernel9<<<(NPIX + 255) / 256, 256, 0, stream>>>(Fraw, S, g2w, g2b, aff, Fkern);

  // ---- 6. w_depth = einsum(gk, unfold(inp_d)) + inp_d ----
  einsum9<<<(NPIX * 4 + 255) / 256, 256, 0, stream>>>(Fkern, S7, S0);   // S0 = w_depth

  // ---- 7. inp_g = res_group(g, gc) ----
  res_group(S3, S7, pk_gc1, gc_c1b, pk_gc2, gc_c2b,
            gc_ca1w, gc_ca1b, gc_ca2w, gc_ca2b, pk_gct, gc_tb);        // S7 = inp_g

  // ---- 8. depth-kernel branch -> kern2 ----
  conv1_64(S0, pk_dk1, dk1b, S4, 2);
  conv1_9(S4, pk_dk2, dk2b, Fraw);
  make_kernel9<<<(NPIX + 255) / 256, 256, 0, stream>>>(Fraw, S, d2w, d2b, aff, Fkern);

  // ---- 9. w_guide = einsum(dk, unfold(inp_g)) + inp_g ----
  einsum9<<<(NPIX * 4 + 255) / 256, 256, 0, stream>>>(Fkern, S7, S1);   // S1 = w_guide

  // ---- 10. output convs (f32 NCHW into d_out) ----
  float* out0 = (float*)d_out;
  float* out1 = out0 + (size_t)BB * CC * HWP;
  conv9(S0, pk_d3, d3b, nullptr, nullptr, out0, 0, 1);
  conv9(S1, pk_g3, g3b, nullptr, nullptr, out1, 0, 1);
}
